// Block_55190329753736
// MI455X (gfx1250) — matrix operand order repaired and checked by reading, not compiled, not run on hardware
//
#include <hip/hip_runtime.h>

// ---------------------------------------------------------------------------
// Transformer block forward (B=2, T=2048, C=1024, H=16, HD=64) for gfx1250.
// GEMMs: v_wmma_f32_16x16x32_bf16, 128x128 block tile, K-step 64,
// double-buffered LDS fed by GLOBAL_LOAD_ASYNC_TO_LDS_B128 (ASYNCcnt path).
// ---------------------------------------------------------------------------

typedef __attribute__((ext_vector_type(16))) __bf16 v16bf;
typedef __attribute__((ext_vector_type(8)))  __bf16 v8bf;
typedef __attribute__((ext_vector_type(8)))  float  v8f;

#if defined(__has_builtin)
#if __has_builtin(__builtin_amdgcn_global_load_async_to_lds_b128)
#define HAVE_ASYNC_LDS 1
#endif
#if __has_builtin(__builtin_amdgcn_s_wait_asynccnt)
#define HAVE_WAIT_ASYNC 1
#endif
#endif

#if defined(HAVE_ASYNC_LDS)
// Builtin signature (from hipcc diagnostic): param0 is
//   '__attribute__((__vector_size__(4*sizeof(int)))) int __device__ *'
// i.e. non-const int4 (gcc vector_size) pointer in AS1; LDS side is AS3.
typedef int v4i_ls __attribute__((vector_size(16)));
typedef __attribute__((address_space(1))) v4i_ls* async_gp;
typedef __attribute__((address_space(3))) v4i_ls* async_lp;
__device__ __forceinline__ void async_b128(const void* g, void* l) {
    __builtin_amdgcn_global_load_async_to_lds_b128((async_gp)g, (async_lp)l, 0, 0);
}
__device__ __forceinline__ void wait_async0() {
#if defined(HAVE_WAIT_ASYNC)
    __builtin_amdgcn_s_wait_asynccnt(0);
#else
    asm volatile("s_wait_asynccnt 0" ::: "memory");
#endif
}
#endif

// Assemble a 16-element bf16 fragment from two contiguous 8-element chunks.
// A-frag:  chunks at K = half*8 and 16 + half*8 (per ISA 16-bit A layout)
// B-frag:  chunks at K = half*8 and 16 + half*8 (same as A-frag)
__device__ __forceinline__ v16bf ld_frag16(const __bf16* p0, const __bf16* p1) {
    v8bf lo = *(const v8bf*)p0;
    v8bf hi = *(const v8bf*)p1;
    v16bf r;
#pragma unroll
    for (int i = 0; i < 8; ++i) { r[i] = lo[i]; r[i + 8] = hi[i]; }
    return r;
}

// ---------------------------------------------------------------------------
// Weight convert + transpose: W[K][N] f32  ->  Wt[N][K] bf16
// ---------------------------------------------------------------------------
__global__ __launch_bounds__(256) void wt_cast_kernel(
    const float* __restrict__ W, __bf16* __restrict__ Wt, int K, int N) {
    __shared__ float tile[32][33];
    const int n0 = blockIdx.x * 32, k0 = blockIdx.y * 32;
    const int tx = threadIdx.x & 31, ty = threadIdx.x >> 5;
#pragma unroll
    for (int i = 0; i < 4; ++i)
        tile[ty + i * 8][tx] = W[(size_t)(k0 + ty + i * 8) * N + n0 + tx];
    __syncthreads();
#pragma unroll
    for (int i = 0; i < 4; ++i)
        Wt[(size_t)(n0 + ty + i * 8) * K + k0 + tx] = (__bf16)tile[tx][ty + i * 8];
}

// ---------------------------------------------------------------------------
// LayerNorm (C=1024) + cast to bf16. One row per 128-thread block.
// ---------------------------------------------------------------------------
__global__ __launch_bounds__(128) void ln_cast_kernel(
    const float* __restrict__ x, const float* __restrict__ w,
    const float* __restrict__ bvec, __bf16* __restrict__ out) {
    const int row = blockIdx.x;
    const int tid = threadIdx.x;
    const float* xr = x + (size_t)row * 1024;
    float v[8];
    *(float4*)&v[0] = *(const float4*)(xr + tid * 8);
    *(float4*)&v[4] = *(const float4*)(xr + tid * 8 + 4);
    float s = 0.f, sq = 0.f;
#pragma unroll
    for (int i = 0; i < 8; ++i) { s += v[i]; sq += v[i] * v[i]; }
#pragma unroll
    for (int m = 16; m >= 1; m >>= 1) {
        s += __shfl_xor(s, m, 32);
        sq += __shfl_xor(sq, m, 32);
    }
    __shared__ float ps[4], pq[4];
    if ((tid & 31) == 0) { ps[tid >> 5] = s; pq[tid >> 5] = sq; }
    __syncthreads();
    s = ps[0] + ps[1] + ps[2] + ps[3];
    sq = pq[0] + pq[1] + pq[2] + pq[3];
    const float mu = s * (1.f / 1024.f);
    const float var = sq * (1.f / 1024.f) - mu * mu;
    const float rstd = rsqrtf(var + 1e-5f);
#pragma unroll
    for (int i = 0; i < 8; ++i) {
        const int c = tid * 8 + i;
        out[(size_t)row * 1024 + c] = (__bf16)((v[i] - mu) * rstd * w[c] + bvec[c]);
    }
}

// ---------------------------------------------------------------------------
// bf16 GEMM:  D[M][N] = A[M][K] * Bt[N][K]^T + bias (+ epilogue)
// Block tile 128x128, K-step 64, 8 waves (256 threads), double-buffered LDS.
// Tiles staged with async global->LDS copies when available (ASYNCcnt),
// else via VGPR staging. Wave grid 4(M) x 2(N); 2x4 accumulators per wave.
// EPI 0: bias -> bf16 out | 1: bias + residual -> f32 out | 2: bias+GELU -> bf16
// ---------------------------------------------------------------------------
template <int EPI>
__global__ __launch_bounds__(256) void gemm_bf16(
    const __bf16* __restrict__ A, const __bf16* __restrict__ Bt,
    const float* __restrict__ bias, const float* __restrict__ resid,
    void* __restrict__ out, int M, int N, int K) {
    __shared__ __align__(16) __bf16 As[2][128][72];
    __shared__ __align__(16) __bf16 Bs[2][128][72];
    const int tid = threadIdx.x;
    const int wave = tid >> 5, lane = tid & 31;
    const int half = lane >> 4, l16 = lane & 15;
    const int wm = wave & 3, wn = wave >> 2;   // 4x2 wave grid
    const int m0 = blockIdx.x * 128;
    const int n0 = blockIdx.y * 128;

    v8f acc[2][4] = {};

    // staging map: thread t -> row = t>>1 (0..127), 32 elems at (t&1)*32
    const int lr = tid >> 1;
    const int lc = (tid & 1) * 32;
    const __bf16* aptr = A + (size_t)(m0 + lr) * K + lc;
    const __bf16* bptr = Bt + (size_t)(n0 + lr) * K + lc;

#if defined(HAVE_ASYNC_LDS)
    // -------- async-to-LDS staging path (no VGPR round trip) --------
#pragma unroll
    for (int i = 0; i < 4; ++i) {
        async_b128(aptr + i * 8, &As[0][lr][lc + i * 8]);
        async_b128(bptr + i * 8, &Bs[0][lr][lc + i * 8]);
    }
    const int nk = K >> 6;
    for (int it = 0; it < nk; ++it) {
        const int buf = it & 1;
        wait_async0();       // my async writes to buf have landed in LDS
        __syncthreads();     // all waves' tiles visible
        if (it + 1 < nk) {   // kick off next tile; overlaps the WMMA block
            const __bf16* ap = aptr + (it + 1) * 64;
            const __bf16* bp = bptr + (it + 1) * 64;
#pragma unroll
            for (int i = 0; i < 4; ++i) {
                async_b128(ap + i * 8, &As[buf ^ 1][lr][lc + i * 8]);
                async_b128(bp + i * 8, &Bs[buf ^ 1][lr][lc + i * 8]);
            }
        }
#pragma unroll
        for (int c = 0; c < 2; ++c) {        // two 32-wide K chunks
            v16bf af[2], bfg[4];
#pragma unroll
            for (int sm = 0; sm < 2; ++sm) {
                const __bf16* arow = &As[buf][wm * 32 + sm * 16 + l16][c * 32];
                af[sm] = ld_frag16(arow + half * 8, arow + 16 + half * 8);
            }
#pragma unroll
            for (int sn = 0; sn < 4; ++sn) {
                const __bf16* brow = &Bs[buf][wn * 64 + sn * 16 + l16][c * 32];
                bfg[sn] = ld_frag16(brow + half * 8, brow + 16 + half * 8);
            }
#pragma unroll
            for (int sm = 0; sm < 2; ++sm)
#pragma unroll
                for (int sn = 0; sn < 4; ++sn)
                    acc[sm][sn] = __builtin_amdgcn_wmma_f32_16x16x32_bf16(
                        false, af[sm], false, bfg[sn], (short)0, acc[sm][sn],
                        false, false);
        }
    }
#else
    // -------- fallback: VGPR staging path --------
    v8bf ar[4], br[4];
#pragma unroll
    for (int i = 0; i < 4; ++i) {
        ar[i] = *(const v8bf*)(aptr + i * 8);
        br[i] = *(const v8bf*)(bptr + i * 8);
    }
#pragma unroll
    for (int i = 0; i < 4; ++i) {
        *(v8bf*)&As[0][lr][lc + i * 8] = ar[i];
        *(v8bf*)&Bs[0][lr][lc + i * 8] = br[i];
    }
    const int nk = K >> 6;
    for (int it = 0; it < nk; ++it) {
        const int buf = it & 1;
        __syncthreads();
        if (it + 1 < nk) {
            const __bf16* ap = aptr + (it + 1) * 64;
            const __bf16* bp = bptr + (it + 1) * 64;
#pragma unroll
            for (int i = 0; i < 4; ++i) {
                ar[i] = *(const v8bf*)(ap + i * 8);
                br[i] = *(const v8bf*)(bp + i * 8);
            }
        }
#pragma unroll
        for (int c = 0; c < 2; ++c) {
            v16bf af[2], bfg[4];
#pragma unroll
            for (int sm = 0; sm < 2; ++sm) {
                const __bf16* arow = &As[buf][wm * 32 + sm * 16 + l16][c * 32];
                af[sm] = ld_frag16(arow + half * 8, arow + 16 + half * 8);
            }
#pragma unroll
            for (int sn = 0; sn < 4; ++sn) {
                const __bf16* brow = &Bs[buf][wn * 64 + sn * 16 + l16][c * 32];
                bfg[sn] = ld_frag16(brow + half * 8, brow + 16 + half * 8);
            }
#pragma unroll
            for (int sm = 0; sm < 2; ++sm)
#pragma unroll
                for (int sn = 0; sn < 4; ++sn)
                    acc[sm][sn] = __builtin_amdgcn_wmma_f32_16x16x32_bf16(
                        false, af[sm], false, bfg[sn], (short)0, acc[sm][sn],
                        false, false);
        }
        if (it + 1 < nk) {
#pragma unroll
            for (int i = 0; i < 4; ++i) {
                *(v8bf*)&As[buf ^ 1][lr][lc + i * 8] = ar[i];
                *(v8bf*)&Bs[buf ^ 1][lr][lc + i * 8] = br[i];
            }
        }
    }
#endif

#pragma unroll
    for (int sm = 0; sm < 2; ++sm) {
        const int row0 = m0 + wm * 32 + sm * 16 + half * 8;
#pragma unroll
        for (int sn = 0; sn < 4; ++sn) {
            const int n = n0 + wn * 64 + sn * 16 + l16;
            const float bv = bias[n];
#pragma unroll
            for (int r = 0; r < 8; ++r) {
                const size_t idx = (size_t)(row0 + r) * N + n;
                float val = acc[sm][sn][r] + bv;
                if (EPI == 2) val = 0.5f * val * (1.0f + erff(val * 0.70710678118654752f));
                if (EPI == 1)
                    ((float*)out)[idx] = resid[idx] + val;
                else
                    ((__bf16*)out)[idx] = (__bf16)val;
            }
        }
    }
}

// ---------------------------------------------------------------------------
// Causal flash attention. Grid: (T/64, H, B); 4 waves, each owns 16 q rows.
// qkv: bf16 [B*T][3C], q|k|v at col offsets 0|C|2C + h*64.
// ---------------------------------------------------------------------------
__global__ __launch_bounds__(128) void attn_kernel(
    const __bf16* __restrict__ qkv, __bf16* __restrict__ y) {
    const int bx = blockIdx.x, h = blockIdx.y, b = blockIdx.z;
    const int tid = threadIdx.x;
    const int wave = tid >> 5, lane = tid & 31, half = lane >> 4, l16 = lane & 15;
    const int C3 = 3072, Cc = 1024, HD = 64, T = 2048;
    const size_t rowbase = (size_t)b * T;
    const int qbase = bx * 64;

    __shared__ __align__(16) __bf16 Qs[64][72];
    __shared__ __align__(16) __bf16 Ks[32][72];
    __shared__ __align__(16) __bf16 Vt[64][40];        // V transposed: [d][key]
    __shared__ __align__(16) __bf16 Ps[4][16][40];     // per-wave P scratch

    {   // stage Q tile (64 rows x 64 dims)
        const int qr = tid >> 1, c0 = (tid & 1) * 32;
        const __bf16* src = qkv + (rowbase + qbase + qr) * C3 + h * HD + c0;
#pragma unroll
        for (int i = 0; i < 4; ++i)
            *(v8bf*)&Qs[qr][c0 + i * 8] = *(const v8bf*)(src + i * 8);
    }
    __syncthreads();

    v16bf qf[2];
    {
        const __bf16* qrow = &Qs[wave * 16 + l16][0];
#pragma unroll
        for (int c = 0; c < 2; ++c)
            qf[c] = ld_frag16(qrow + c * 32 + half * 8, qrow + c * 32 + 16 + half * 8);
    }

    float mi[8], li[8];
    v8f of[4] = {};
#pragma unroll
    for (int r = 0; r < 8; ++r) { mi[r] = -1e30f; li[r] = 0.f; }

    const int nkt = 2 * bx + 2;
    for (int kt = 0; kt < nkt; ++kt) {
        const int kbase = kt * 32;
        __syncthreads();
        {   // stage K tile (32x64) and V^T tile (64x32)
            const int kr = tid >> 2, c0 = (tid & 3) * 16;
            const __bf16* ksrc = qkv + (rowbase + kbase + kr) * C3 + Cc + h * HD + c0;
            *(v8bf*)&Ks[kr][c0] = *(const v8bf*)ksrc;
            *(v8bf*)&Ks[kr][c0 + 8] = *(const v8bf*)(ksrc + 8);
            const __bf16* vsrc = qkv + (rowbase + kbase + kr) * C3 + 2 * Cc + h * HD + c0;
#pragma unroll
            for (int i = 0; i < 16; ++i) Vt[c0 + i][kr] = vsrc[i];
        }
        __syncthreads();

        // S = Q K^T  (16 q rows x 32 keys per wave)
        v8f sacc[2] = {};
#pragma unroll
        for (int c = 0; c < 2; ++c) {
#pragma unroll
            for (int sub = 0; sub < 2; ++sub) {
                const __bf16* krow = &Ks[sub * 16 + l16][c * 32];
                const v16bf kf = ld_frag16(krow + half * 8, krow + 16 + half * 8);
                sacc[sub] = __builtin_amdgcn_wmma_f32_16x16x32_bf16(
                    false, qf[c], false, kf, (short)0, sacc[sub], false, false);
            }
        }

        // scale + causal mask + online softmax
        float mnew[8];
#pragma unroll
        for (int r = 0; r < 8; ++r) {
            const int rowg = qbase + wave * 16 + half * 8 + r;
#pragma unroll
            for (int sub = 0; sub < 2; ++sub) {
                const int kg = kbase + sub * 16 + l16;
                const float s = sacc[sub][r] * 0.125f;   // 1/sqrt(64)
                sacc[sub][r] = (kg <= rowg) ? s : -1e30f;
            }
            float m = fmaxf(sacc[0][r], sacc[1][r]);
            m = fmaxf(m, __shfl_xor(m, 1, 32));
            m = fmaxf(m, __shfl_xor(m, 2, 32));
            m = fmaxf(m, __shfl_xor(m, 4, 32));
            m = fmaxf(m, __shfl_xor(m, 8, 32));
            mnew[r] = fmaxf(mi[r], m);
        }
#pragma unroll
        for (int r = 0; r < 8; ++r) {
            const float alpha = __expf(mi[r] - mnew[r]);
            float rsum = 0.f;
#pragma unroll
            for (int sub = 0; sub < 2; ++sub) {
                const float p = __expf(sacc[sub][r] - mnew[r]);
                sacc[sub][r] = p;
                rsum += p;
            }
            rsum += __shfl_xor(rsum, 1, 32);
            rsum += __shfl_xor(rsum, 2, 32);
            rsum += __shfl_xor(rsum, 4, 32);
            rsum += __shfl_xor(rsum, 8, 32);
            li[r] = li[r] * alpha + rsum;
            mi[r] = mnew[r];
#pragma unroll
            for (int f = 0; f < 4; ++f) of[f][r] *= alpha;
        }

        // P (C layout) -> per-wave LDS -> A-fragment layout (bf16)
#pragma unroll
        for (int sub = 0; sub < 2; ++sub)
#pragma unroll
            for (int r = 0; r < 8; ++r)
                Ps[wave][half * 8 + r][sub * 16 + l16] = (__bf16)sacc[sub][r];
        asm volatile("" ::: "memory");   // same-wave LDS is in-order; block reorder
        const __bf16* prow = &Ps[wave][l16][0];
        const v16bf pf = ld_frag16(prow + half * 8, prow + 16 + half * 8);

        // O += P V
#pragma unroll
        for (int f = 0; f < 4; ++f) {
            const __bf16* vrow = &Vt[f * 16 + l16][0];
            const v16bf vf = ld_frag16(vrow + half * 8, vrow + 16 + half * 8);
            of[f] = __builtin_amdgcn_wmma_f32_16x16x32_bf16(
                false, pf, false, vf, (short)0, of[f], false, false);
        }
    }

    // normalize + write y (bf16, [B*T][C])
#pragma unroll
    for (int f = 0; f < 4; ++f) {
        const int col = h * HD + f * 16 + l16;
#pragma unroll
        for (int r = 0; r < 8; ++r) {
            const int rowg = qbase + wave * 16 + half * 8 + r;
            y[(rowbase + rowg) * 1024 + col] = (__bf16)(of[f][r] / li[r]);
        }
    }
}

// ---------------------------------------------------------------------------
// Orchestration
// ---------------------------------------------------------------------------
extern "C" void kernel_launch(void* const* d_in, const int* in_sizes, int n_in,
                              void* d_out, int out_size, void* d_ws, size_t ws_size,
                              hipStream_t stream) {
    const float* x        = (const float*)d_in[0];
    const float* ln1_w    = (const float*)d_in[1];
    const float* ln1_b    = (const float*)d_in[2];
    const float* attn_w   = (const float*)d_in[3];
    const float* attn_b   = (const float*)d_in[4];
    const float* proj_w   = (const float*)d_in[5];
    const float* proj_b   = (const float*)d_in[6];
    const float* ln2_w    = (const float*)d_in[7];
    const float* ln2_b    = (const float*)d_in[8];
    const float* fc_w     = (const float*)d_in[9];
    const float* fc_b     = (const float*)d_in[10];
    const float* fcp_w    = (const float*)d_in[11];
    const float* fcp_b    = (const float*)d_in[12];

    const int M = 4096;  // B*T
    char* ws = (char*)d_ws;
    size_t off = 0;
    auto carve = [&](size_t bytes) { char* p = ws + off; off += (bytes + 255) & ~(size_t)255; return p; };
    __bf16* attn_wt = (__bf16*)carve((size_t)3072 * 1024 * 2);
    __bf16* proj_wt = (__bf16*)carve((size_t)1024 * 1024 * 2);
    __bf16* fc_wt   = (__bf16*)carve((size_t)4096 * 1024 * 2);
    __bf16* fcp_wt  = (__bf16*)carve((size_t)1024 * 4096 * 2);
    __bf16* h1      = (__bf16*)carve((size_t)M * 1024 * 2);
    __bf16* qkv     = (__bf16*)carve((size_t)M * 3072 * 2);
    __bf16* yattn   = (__bf16*)carve((size_t)M * 1024 * 2);
    float*  x1      = (float*) carve((size_t)M * 1024 * 4);
    __bf16* h2      = (__bf16*)carve((size_t)M * 1024 * 2);
    __bf16* hg      = (__bf16*)carve((size_t)M * 4096 * 2);

    // weights -> transposed bf16
    wt_cast_kernel<<<dim3(3072 / 32, 1024 / 32), 256, 0, stream>>>(attn_w, attn_wt, 1024, 3072);
    wt_cast_kernel<<<dim3(1024 / 32, 1024 / 32), 256, 0, stream>>>(proj_w, proj_wt, 1024, 1024);
    wt_cast_kernel<<<dim3(4096 / 32, 1024 / 32), 256, 0, stream>>>(fc_w, fc_wt, 1024, 4096);
    wt_cast_kernel<<<dim3(1024 / 32, 4096 / 32), 256, 0, stream>>>(fcp_w, fcp_wt, 4096, 1024);

    // LN1 -> h1 (bf16)
    ln_cast_kernel<<<M, 128, 0, stream>>>(x, ln1_w, ln1_b, h1);
    // qkv = h1 @ attn_w + attn_b   [4096 x 3072]
    gemm_bf16<0><<<dim3(M / 128, 3072 / 128), 256, 0, stream>>>(
        h1, attn_wt, attn_b, nullptr, qkv, M, 3072, 1024);
    // attention -> yattn (bf16)
    attn_kernel<<<dim3(2048 / 64, 16, 2), 128, 0, stream>>>(qkv, yattn);
    // x1 = x + yattn @ proj_w + proj_b   (f32)
    gemm_bf16<1><<<dim3(M / 128, 1024 / 128), 256, 0, stream>>>(
        yattn, proj_wt, proj_b, x, x1, M, 1024, 1024);
    // LN2 -> h2 (bf16)
    ln_cast_kernel<<<M, 128, 0, stream>>>(x1, ln2_w, ln2_b, h2);
    // hg = gelu(h2 @ fc_w + fc_b)   [4096 x 4096]
    gemm_bf16<2><<<dim3(M / 128, 4096 / 128), 256, 0, stream>>>(
        h2, fc_wt, fc_b, nullptr, hg, M, 4096, 1024);
    // out = x1 + hg @ fc_proj_w + fc_proj_b   (f32)
    gemm_bf16<1><<<dim3(M / 128, 1024 / 128), 256, 0, stream>>>(
        hg, fcp_wt, fcp_b, x1, (float*)d_out, M, 1024, 4096);
}
